// kernel_generated_1_62904091017450
// MI455X (gfx1250) — matrix operand order repaired and checked by reading, not compiled, not run on hardware
//
#include <hip/hip_runtime.h>

// ---------------------------------------------------------------------------
// out[m,i,p,o] = sum_{kh,c,kw} W[i,kh,c,kw] * u[m,kh,p,c,o+kw-1]   (pad in o)
//   u[m,kh,p,c,j] = [h2=p+kh-1 in range]      x[m, g2%28, g2/28, j], g2=h2*64+c
//                 + [h1=(p?p+kh-2:kh+26) ok]  x[m, g1%28, g1/28, j], g1=h1*64+32+c
// Batched GEMM per sample m: M=128 (i), N=784 (p*28+o), K=288 (kh*96+kw*32+c).
// bf16 WMMA, f32 accumulate. Block = 1 sample x 112 columns (784 = 7*112),
// 8 waves x (M=16, N=112 -> 7 accumulators), 9 K-steps of 32 -> 63 wmma/wave.
// ---------------------------------------------------------------------------

typedef __bf16 bf16_t;
typedef __attribute__((ext_vector_type(16))) __bf16 v16bf;
typedef __attribute__((ext_vector_type(8)))  float  v8f;

#define NCOLS 112
#define KTOT  288   // row stride in Bt (no pad: 576B rows stay 16B aligned)

// XOR-swizzle of 16B chunks inside a row: kills the stride-288 (=16 dwords
// mod 64) 4-way LDS bank conflict on ds_load_b128 fragment reads.
__device__ __forceinline__ int bt_addr(int n, int k) {
  int chunk = (k >> 3) ^ ((n >> 2) & 3);
  return n * KTOT + (chunk << 3) + (k & 7);
}

// Pre-kernel: W fp32 [i][kh][c][kw] -> bf16 [i][kh*96 + kw*32 + c] in d_ws.
__global__ __launch_bounds__(256) void wconv_bf16(const float* __restrict__ W,
                                                  bf16_t* __restrict__ wb) {
  int idx = blockIdx.x * 256 + threadIdx.x;     // [0, 36864)
  int i  = idx / 288;
  int k  = idx - i * 288;
  int kh = k / 96;
  int r  = k - kh * 96;
  int kw = r >> 5;
  int c  = r & 31;
  wb[idx] = (bf16_t)W[((i * 3 + kh) * 32 + c) * 3 + kw];
}

__global__ __launch_bounds__(256) void fused_shiftconv(const float* __restrict__ x,
                                                       const bf16_t* __restrict__ wb,
                                                       float* __restrict__ out) {
  __shared__ bf16_t Bt[NCOLS * KTOT];            // 112 x 288 bf16 = 64,512 B
  const int tid = threadIdx.x;
  const int m   = blockIdx.y;                    // sample
  const int n0  = blockIdx.x * NCOLS;            // column-tile base
  const float* xs = x + (size_t)m * 50176;       // x[m][k][i][j]: strides 1792/28/1

  // ---- pack stage: build u-derived B^T tile (112 cols x 288 K) in LDS ----
  // consecutive lanes -> consecutive columns -> consecutive o -> coalesced x reads
  #pragma unroll 2
  for (int e = 0; e < (NCOLS * KTOT) / 256; ++e) {   // 126 iterations
    int idx = e * 256 + tid;             // [0, 32256)
    int k   = idx / NCOLS;               // [0, 288)
    int n   = idx - k * NCOLS;           // [0, 112)
    int col = n0 + n;                    // [0, 784)
    int p   = col / 28;
    int o   = col - p * 28;
    int kh  = k / 96;
    int r   = k - kh * 96;
    int kw  = r >> 5;
    int c   = r & 31;
    int oo  = o + kw - 1;                // second unfold (pad 1)
    float v = 0.0f;
    if (oo >= 0 && oo < 28) {
      int h2 = p + kh - 1;                              // q=1 term (even s)
      if ((unsigned)h2 < 28u) {
        int g = h2 * 64 + c;
        v += xs[(g % 28) * 1792 + (g / 28) * 28 + oo];
      }
      int h1 = (p == 0) ? (kh + 26) : (p + kh - 2);     // q=0 term (odd s, roll wrap)
      if ((unsigned)h1 < 28u) {
        int g = h1 * 64 + 32 + c;
        v += xs[(g % 28) * 1792 + (g / 28) * 28 + oo];
      }
    }
    Bt[bt_addr(n, k)] = (bf16_t)v;
  }
  __syncthreads();

  // ---- GEMM stage: wave w -> channels [16w,16w+16); 7 column-tiles reuse A ----
  const int wave  = tid >> 5;
  const int lane  = tid & 31;
  const int row16 = lane & 15;
  const int khalf = lane >> 4;

  const bf16_t* arow = wb + (size_t)((wave << 4) + row16) * 288;  // A row (L2)
  const int akoff = khalf << 3;   // A: lanes 0-15 hold K 0-7/16-23; 16-31: 8-15/24-31
  const int bkoff = khalf << 3;   // B: lanes 0-15 hold K 0-7/16-23; 16-31: 8-15/24-31

  v8f acc[7];
  #pragma unroll
  for (int t = 0; t < 7; ++t) acc[t] = (v8f){};

  #pragma unroll
  for (int ks = 0; ks < 9; ++ks) {
    const int kb = ks << 5;
    v16bf a;
    ((uint4*)&a)[0] = *(const uint4*)(arow + kb + akoff);
    ((uint4*)&a)[1] = *(const uint4*)(arow + kb + akoff + 16);
    #pragma unroll
    for (int t = 0; t < 7; ++t) {
      const int bn = (t << 4) + row16;             // B^T row (column index)
      v16bf b;
      ((uint4*)&b)[0] = *(const uint4*)(Bt + bt_addr(bn, kb + bkoff));
      ((uint4*)&b)[1] = *(const uint4*)(Bt + bt_addr(bn, kb + bkoff + 16));
      acc[t] = __builtin_amdgcn_wmma_f32_16x16x32_bf16(
          /*neg_a=*/false, a, /*neg_b=*/false, b,
          /*c_mod=*/(short)0, acc[t], /*reuse_a=*/false, /*reuse_b=*/false);
    }
  }

  // C/D layout: VGPR r -> M = khalf*8 + r, N = row16 (per 16-col tile t)
  float* op = out + (size_t)m * 100352
                  + (size_t)((wave << 4) + (khalf << 3)) * 784
                  + (n0 + row16);
  #pragma unroll
  for (int t = 0; t < 7; ++t) {
    #pragma unroll
    for (int r2 = 0; r2 < 8; ++r2) op[(t << 4) + r2 * 784] = acc[t][r2];
  }
}

extern "C" void kernel_launch(void* const* d_in, const int* in_sizes, int n_in,
                              void* d_out, int out_size, void* d_ws, size_t ws_size,
                              hipStream_t stream) {
  (void)in_sizes; (void)n_in; (void)out_size; (void)ws_size;
  const float* x = (const float*)d_in[0];   // (1024, 28, 64, 28) fp32
  const float* W = (const float*)d_in[1];   // (128, 3, 32, 3)    fp32
  float* out = (float*)d_out;               // (1024, 128, 28, 28) fp32
  bf16_t* wb = (bf16_t*)d_ws;               // 36,864 bf16 = 73,728 B scratch

  wconv_bf16<<<144, 256, 0, stream>>>(W, wb);
  dim3 grid(7, 1024);                       // 7 column-tiles x 1024 samples
  fused_shiftconv<<<grid, 256, 0, stream>>>(x, wb, out);
}
